// AttentionDiT_26714696581157
// MI455X (gfx1250) — compile-verified, matrix operand order checked by reading, not run on hardware
//
#include <hip/hip_runtime.h>
#include <hip/hip_bf16.h>

// ---------------------------------------------------------------------------
// DiT attention block on gfx1250 (wave32, WMMA f32_16x16x32_f16)
//   K1: qkv = x @ W_qkv           -> scatter f16 Q/K/V [3][B*H][N][64] in ws
//   K2: flash attention per head  -> attn_out f16 [B, N, 768] in ws
//   K3: out = attn_out @ W_proj+b -> f32 d_out
// Async GLOBAL->LDS (ASYNCcnt) used for pure-copy tile fills when available.
// ---------------------------------------------------------------------------

typedef __attribute__((ext_vector_type(16))) _Float16 v16h;
typedef __attribute__((ext_vector_type(8)))  _Float16 v8h;
typedef __attribute__((ext_vector_type(4)))  _Float16 v4h;
typedef __attribute__((ext_vector_type(8)))  float    v8f;
typedef int v4i __attribute__((vector_size(16)));  // matches async-LDS builtin pointee

union Frag16 { v16h v; v8h h[2]; };

#if defined(__AMDGCN__) && __has_builtin(__builtin_amdgcn_global_load_async_to_lds_b128)
#define USE_ASYNC_LDS 1
#else
#define USE_ASYNC_LDS 0
#endif

#if USE_ASYNC_LDS && __has_builtin(__builtin_amdgcn_s_wait_asynccnt)
#define ASYNC_WAIT() __builtin_amdgcn_s_wait_asynccnt(0)
#elif USE_ASYNC_LDS
#define ASYNC_WAIT() asm volatile("s_wait_asynccnt 0" ::: "memory")
#else
#define ASYNC_WAIT()
#endif

// Generic->AS1 / AS3 via integer round trip (LDS offset lives in the low 32
// bits of a generic LDS address per the flat-aperture rules). Builtin wants
// int4-vector pointees.
#define AS1P(p) ((__attribute__((address_space(1))) v4i*)(unsigned long long)(p))
#define AS3P(p) ((__attribute__((address_space(3))) v4i*)(unsigned int)(unsigned long long)(p))

__device__ __forceinline__ void async_b128(const _Float16* g, _Float16* l) {
#if USE_ASYNC_LDS
  // per-lane 16B: LDS[l..l+15] = MEM[g..g+15], tracked by ASYNCcnt
  __builtin_amdgcn_global_load_async_to_lds_b128(AS1P(g), AS3P(l), 0, 0);
#else
  (void)g; (void)l;
#endif
}

__device__ __forceinline__ v8f wmma_f16(const v16h& a, const v16h& b, const v8f& c) {
  // 8 args: (neg_a, A, neg_b, B, c_mod, C, reuse_a, reuse_b)
  return __builtin_amdgcn_wmma_f32_16x16x32_f16(false, a, false, b, (short)0, c,
                                                false, false);
}

__device__ __forceinline__ v8f vzero8() {
  v8f z;
#pragma unroll
  for (int i = 0; i < 8; ++i) z[i] = 0.0f;
  return z;
}

// ---------------------------------------------------------------------------
// GEMM: C[M,N] = A[M,K] * W[K,N]
//   MODE 0: A = f32 (x), scatter C as f16 into qkv buffers [3][24][4096][64]
//   MODE 1: A = f16 (attn out), C = f32 out + bias  (A fill via async LDS)
// Block tile 128x128, k-chunk 32, 8 waves (2x4), wave tile 64x32.
// LDS holds A-tile row-major [m][k] and W-tile TRANSPOSED [n][k] so both
// WMMA fragments load as two ds_load_b128 per 16x32 fragment.
// ---------------------------------------------------------------------------
template <int MODE>
__global__ __launch_bounds__(256) void gemm_wmma(
    const float* __restrict__ A32, const _Float16* __restrict__ A16,
    const float* __restrict__ W, const float* __restrict__ bias,
    _Float16* __restrict__ qkv, float* __restrict__ out,
    int M, int K, int N) {
  constexpr int BM = 128, BN = 128, BK = 32, LDA = 40;  // 40-half rows: 80B, 16B-aligned
  __shared__ __align__(16) _Float16 As[2][BM * LDA];
  __shared__ __align__(16) _Float16 Bs[2][BN * LDA];

  const int tid = threadIdx.x;
  const int lane = tid & 31;
  const int wm = (tid >> 5) >> 2;  // 0..1  (64 rows each)
  const int wn = (tid >> 5) & 3;   // 0..3  (32 cols each)
  const int m0 = blockIdx.y * BM;
  const int n0 = blockIdx.x * BN;
  const int KC = K / BK;
  const int l15 = lane & 15;
  const int koff = ((lane >> 4) & 1) * 8;

  auto fill = [&](int buf, int kc) {
    {  // A tile: [128 m][32 k]
      const int r = tid >> 1;
      const int c = (tid & 1) * 16;
      if (MODE == 0) {
        const float* g = A32 + (size_t)(m0 + r) * K + kc * BK + c;
#pragma unroll
        for (int i = 0; i < 4; ++i) {
          const float4 f = *(const float4*)(g + i * 4);
          v4h hv;
          hv[0] = (_Float16)f.x; hv[1] = (_Float16)f.y;
          hv[2] = (_Float16)f.z; hv[3] = (_Float16)f.w;
          *(v4h*)&As[buf][r * LDA + c + i * 4] = hv;
        }
      } else {
        const _Float16* g = A16 + (size_t)(m0 + r) * K + kc * BK + c;
        _Float16* l = &As[buf][r * LDA + c];
        if (USE_ASYNC_LDS) {
          async_b128(g, l);
          async_b128(g + 8, l + 8);
        } else {
          *(v8h*)l       = *(const v8h*)(g);
          *(v8h*)(l + 8) = *(const v8h*)(g + 8);
        }
      }
    }
    {  // W tile transposed: Bs[n][k] = W[kc*32+k][n0+n]
      const int k = tid >> 3;         // 0..31
      const int nb = (tid & 7) * 16;  // 0..112
      const float* g = W + (size_t)(kc * BK + k) * N + n0 + nb;
#pragma unroll
      for (int i = 0; i < 4; ++i) {
        const float4 f = *(const float4*)(g + i * 4);
        Bs[buf][(nb + i * 4 + 0) * LDA + k] = (_Float16)f.x;
        Bs[buf][(nb + i * 4 + 1) * LDA + k] = (_Float16)f.y;
        Bs[buf][(nb + i * 4 + 2) * LDA + k] = (_Float16)f.z;
        Bs[buf][(nb + i * 4 + 3) * LDA + k] = (_Float16)f.w;
      }
    }
  };

  v8f acc[4][2];
#pragma unroll
  for (int mt = 0; mt < 4; ++mt)
#pragma unroll
    for (int nt = 0; nt < 2; ++nt) acc[mt][nt] = vzero8();

  fill(0, 0);
  for (int kc = 0; kc < KC; ++kc) {
    if (MODE == 1) ASYNC_WAIT();
    __syncthreads();
    if (kc + 1 < KC) fill((kc + 1) & 1, kc + 1);
    const int buf = kc & 1;

    Frag16 a[4], b[2];
#pragma unroll
    for (int mt = 0; mt < 4; ++mt) {
      const _Float16* p = &As[buf][(wm * 64 + mt * 16 + l15) * LDA + koff];
      a[mt].h[0] = *(const v8h*)p;
      a[mt].h[1] = *(const v8h*)(p + 16);
    }
#pragma unroll
    for (int nt = 0; nt < 2; ++nt) {
      const _Float16* p = &Bs[buf][(wn * 32 + nt * 16 + l15) * LDA + koff];
      b[nt].h[0] = *(const v8h*)p;
      b[nt].h[1] = *(const v8h*)(p + 16);
    }
#pragma unroll
    for (int mt = 0; mt < 4; ++mt)
#pragma unroll
      for (int nt = 0; nt < 2; ++nt)
        acc[mt][nt] = wmma_f16(a[mt].v, b[nt].v, acc[mt][nt]);
  }

  // Epilogue. C layout: row m = g + 8*(lane>=16), col n = lane&15 per tile.
#pragma unroll
  for (int mt = 0; mt < 4; ++mt) {
#pragma unroll
    for (int nt = 0; nt < 2; ++nt) {
#pragma unroll
      for (int g = 0; g < 8; ++g) {
        const int m = m0 + wm * 64 + mt * 16 + g + ((lane >> 4) << 3);
        const int n = n0 + wn * 32 + nt * 16 + l15;
        const float val = acc[mt][nt][g];
        if (MODE == 0) {
          const int which = n / 768;             // 0:q 1:k 2:v
          const int cc = n - which * 768;
          const int h = cc >> 6, d = cc & 63;    // head, dim
          const int bb = m >> 12, nn = m & 4095; // batch, token
          qkv[(((size_t)which * 24 + bb * 12 + h) * 4096 + nn) * 64 + d] =
              (_Float16)val;
        } else {
          out[(size_t)m * N + n] = val + bias[n];
        }
      }
    }
  }
}

// ---------------------------------------------------------------------------
// Flash attention, one (b,h) head slice + 128 query rows per block.
// 8 waves; each wave owns 16 query rows. Keys processed in 64-wide chunks,
// K (row-major [key][d], async-filled) and V (transposed [d][key]) double-
// buffered in LDS. Online softmax in the WMMA C-layout; the softmax scale is
// pre-folded into the f16 Q fragments (exact: 0.125). P goes through per-wave
// LDS to become an f16 A-fragment for the P*V accumulation.
// ---------------------------------------------------------------------------
__global__ __launch_bounds__(256) void attn_wmma(
    const _Float16* __restrict__ Q, const _Float16* __restrict__ Kmat,
    const _Float16* __restrict__ V, _Float16* __restrict__ O) {
  constexpr int D = 64, NKEY = 4096, NC = 64, LDK = 72;  // 72 halves = 144B rows
  __shared__ __align__(16) _Float16 Ks[2][NC * LDK];   // [key][d]
  __shared__ __align__(16) _Float16 Vs[2][D * LDK];    // [d][key] (transposed)
  __shared__ __align__(16) _Float16 Ps[8][16 * 64];    // per-wave P staging

  const int tid = threadIdx.x, lane = tid & 31, wv = tid >> 5;
  const int bh = blockIdx.x >> 5;  // 0..23 = b*12+h
  const int qb = blockIdx.x & 31;  // query block 0..31
  const int b = bh / 12, h = bh - b * 12;
  const _Float16* Qh = Q + (size_t)bh * NKEY * D;
  const _Float16* Kh = Kmat + (size_t)bh * NKEY * D;
  const _Float16* Vh = V + (size_t)bh * NKEY * D;
  const int q0 = qb * 128 + wv * 16;
  const int l15 = lane & 15;
  const int koff = ((lane >> 4) & 1) * 8;

  // Q fragments live in registers the whole kernel; fold in softmax scale.
  Frag16 aq[2];
#pragma unroll
  for (int kk = 0; kk < 2; ++kk) {
    const _Float16* p = Qh + (size_t)(q0 + l15) * D + kk * 32 + koff;
    aq[kk].h[0] = *(const v8h*)p;
    aq[kk].h[1] = *(const v8h*)(p + 16);
#pragma unroll
    for (int e = 0; e < 16; ++e)
      aq[kk].v[e] = aq[kk].v[e] * (_Float16)0.125f;  // 64^-0.5, exact in f16
  }

  v8f acc[4];
#pragma unroll
  for (int dt = 0; dt < 4; ++dt) acc[dt] = vzero8();
  float mrow[8], lrow[8];
#pragma unroll
  for (int g = 0; g < 8; ++g) { mrow[g] = -3.0e38f; lrow[g] = 0.0f; }

  auto fill = [&](int buf, int j0) {
    const int r = tid >> 2;         // key 0..63
    const int cb = (tid & 3) * 16;  // d 0..48
    const _Float16* kg = Kh + (size_t)(j0 + r) * D + cb;
    _Float16* kl = &Ks[buf][r * LDK + cb];
    if (USE_ASYNC_LDS) {
      async_b128(kg, kl);
      async_b128(kg + 8, kl + 8);
    } else {
      *(v8h*)kl       = *(const v8h*)kg;
      *(v8h*)(kl + 8) = *(const v8h*)(kg + 8);
    }
    const _Float16* vg = Vh + (size_t)(j0 + r) * D + cb;
    const v8h v0 = *(const v8h*)vg;
    const v8h v1 = *(const v8h*)(vg + 8);
#pragma unroll
    for (int i = 0; i < 8; ++i) {
      Vs[buf][(cb + i) * LDK + r]     = v0[i];
      Vs[buf][(cb + 8 + i) * LDK + r] = v1[i];
    }
  };

  fill(0, 0);
  for (int jc = 0; jc < NKEY / NC; ++jc) {
    ASYNC_WAIT();
    __syncthreads();
    if (jc + 1 < NKEY / NC) fill((jc + 1) & 1, (jc + 1) * NC);
    const int buf = jc & 1;

    // S = (Q*scale) K^T for 64 keys: 4 n-tiles x 2 k-chunks of WMMA.
    v8f S[4];
#pragma unroll
    for (int nt = 0; nt < 4; ++nt) {
      v8f s = vzero8();
#pragma unroll
      for (int kk = 0; kk < 2; ++kk) {
        Frag16 bk;
        const _Float16* p = &Ks[buf][(nt * 16 + l15) * LDK + kk * 32 + koff];
        bk.h[0] = *(const v8h*)p;
        bk.h[1] = *(const v8h*)(p + 16);
        s = wmma_f16(aq[kk].v, bk.v, s);
      }
      S[nt] = s;
    }

    // Online softmax. Rows of C-layout live on 16-lane halves: reduce with
    // xor-shuffles (masks 1/2/4/8 stay inside the half).
    float corr[8];
#pragma unroll
    for (int g = 0; g < 8; ++g) {
      float mx = -3.0e38f;
#pragma unroll
      for (int nt = 0; nt < 4; ++nt) mx = fmaxf(mx, S[nt][g]);
#pragma unroll
      for (int sh = 1; sh < 16; sh <<= 1) mx = fmaxf(mx, __shfl_xor(mx, sh, 32));
      const float mnew = fmaxf(mrow[g], mx);
      corr[g] = __expf(mrow[g] - mnew);
      float sum = 0.0f;
#pragma unroll
      for (int nt = 0; nt < 4; ++nt) {
        const float e = __expf(S[nt][g] - mnew);
        S[nt][g] = e;
        sum += e;
      }
#pragma unroll
      for (int sh = 1; sh < 16; sh <<= 1) sum += __shfl_xor(sum, sh, 32);
      mrow[g] = mnew;
      lrow[g] = lrow[g] * corr[g] + sum;
    }

    // P -> per-wave LDS (C-layout scatter), then reload as A-fragments.
    _Float16* Pw = &Ps[wv][0];
#pragma unroll
    for (int nt = 0; nt < 4; ++nt)
#pragma unroll
      for (int g = 0; g < 8; ++g)
        Pw[(g + ((lane >> 4) << 3)) * 64 + nt * 16 + l15] = (_Float16)S[nt][g];

#pragma unroll
    for (int dt = 0; dt < 4; ++dt)
#pragma unroll
      for (int g = 0; g < 8; ++g) acc[dt][g] *= corr[g];

    Frag16 ap[2];
#pragma unroll
    for (int kk = 0; kk < 2; ++kk) {
      const _Float16* p = &Pw[l15 * 64 + kk * 32 + koff];
      ap[kk].h[0] = *(const v8h*)p;
      ap[kk].h[1] = *(const v8h*)(p + 16);
    }

    // O += P * V (V stored d-major so B-fragments are contiguous b128 loads).
#pragma unroll
    for (int dt = 0; dt < 4; ++dt) {
#pragma unroll
      for (int kk = 0; kk < 2; ++kk) {
        Frag16 bv;
        const _Float16* p = &Vs[buf][(dt * 16 + l15) * LDK + kk * 32 + koff];
        bv.h[0] = *(const v8h*)p;
        bv.h[1] = *(const v8h*)(p + 16);
        acc[dt] = wmma_f16(ap[kk].v, bv.v, acc[dt]);
      }
    }
  }

  // Normalize and store f16 to attn_out [B, N, 768].
  float rinv[8];
#pragma unroll
  for (int g = 0; g < 8; ++g) rinv[g] = 1.0f / lrow[g];
#pragma unroll
  for (int dt = 0; dt < 4; ++dt) {
#pragma unroll
    for (int g = 0; g < 8; ++g) {
      const int qrow = q0 + g + ((lane >> 4) << 3);
      const float o = acc[dt][g] * rinv[g];
      O[(size_t)(b * 4096 + qrow) * 768 + h * 64 + dt * 16 + l15] = (_Float16)o;
    }
  }
}

// ---------------------------------------------------------------------------
extern "C" void kernel_launch(void* const* d_in, const int* in_sizes, int n_in,
                              void* d_out, int out_size, void* d_ws,
                              size_t ws_size, hipStream_t stream) {
  (void)in_sizes; (void)n_in; (void)out_size; (void)ws_size;
  const float* x      = (const float*)d_in[0];
  const float* W_qkv  = (const float*)d_in[1];
  const float* W_proj = (const float*)d_in[2];
  const float* b_proj = (const float*)d_in[3];

  const int M = 2 * 4096;  // 8192 tokens
  const int C = 768;
  const size_t per = (size_t)24 * 4096 * 64;  // halves per q/k/v tensor

  _Float16* qkvbuf = (_Float16*)d_ws;   // [3][24][4096][64] f16
  _Float16* Qb = qkvbuf;
  _Float16* Kb = qkvbuf + per;
  _Float16* Vb = qkvbuf + 2 * per;
  _Float16* AO = qkvbuf + 3 * per;      // [8192][768] f16

  gemm_wmma<0><<<dim3((3 * C) / 128, M / 128), 256, 0, stream>>>(
      x, nullptr, W_qkv, nullptr, qkvbuf, nullptr, M, C, 3 * C);

  attn_wmma<<<dim3(24 * 32), 256, 0, stream>>>(Qb, Kb, Vb, AO);

  gemm_wmma<1><<<dim3(C / 128, M / 128), 256, 0, stream>>>(
      nullptr, AO, W_proj, b_proj, nullptr, (float*)d_out, M, C, C);
}
